// GluTracker_5102421148400
// MI455X (gfx1250) — matrix operand order repaired and checked by reading, not compiled, not run on hardware
//
#include <hip/hip_runtime.h>

#define C_DIM 128
#define H_DIM 108
#define W_DIM 135
#define P49   49          // 7x7 window positions
#define SSTRIDE 132       // f32 staging row stride (pad vs 128 for LDS banks)
#define ASTRIDE 136       // f16 A row stride (16B-aligned: 136*2=272=16*17)
#define BSTRIDE 72        // f16 B row stride (16B-aligned: 72*2=144=16*9)
#define DSTRIDE 65        // f32 D-staging row stride (64 cols + 1 pad)

typedef __attribute__((ext_vector_type(16))) _Float16 v16h;
typedef __attribute__((ext_vector_type(8)))  _Float16 v8h;
typedef __attribute__((ext_vector_type(8)))  float    v8f;

// Load 8 horizontally-contiguous pixels of one channel plane with zero padding.
__device__ __forceinline__ void load_row(const float* __restrict__ plane,
                                         int Y, int Xs, float* r) {
    if (Y < 0 || Y >= H_DIM) {
        #pragma unroll
        for (int i = 0; i < 8; ++i) r[i] = 0.0f;
        return;
    }
    const float* rp = plane + Y * W_DIM;
    #pragma unroll
    for (int i = 0; i < 8; ++i) {
        int X = Xs + i;
        r[i] = (X >= 0 && X < W_DIM) ? rp[X] : 0.0f;
    }
}

// Thread c samples its channel's 7x7 bilinear window into staging S[pos][ch].
// All 49 window positions share one fractional offset (integer window offsets),
// so an 8x8 integer patch + fixed 4 weights reproduces the reference exactly.
__device__ __forceinline__ void sample_to_S(const float* __restrict__ fmap,
                                            float px, float py, int c,
                                            float* __restrict__ S) {
    float fx = px * (float)(W_DIM - 1);
    float fy = py * (float)(H_DIM - 1);
    float x0f = floorf(fx), y0f = floorf(fy);
    int x0 = (int)x0f, y0 = (int)y0f;
    float wx1 = fx - x0f, wx0 = 1.0f - wx1;
    float wy1 = fy - y0f, wy0 = 1.0f - wy1;
    const float* plane = fmap + c * (H_DIM * W_DIM);
    float rA[8], rB[8];
    load_row(plane, y0 - 3, x0 - 3, rA);
    #pragma unroll
    for (int dy = 0; dy < 7; ++dy) {
        load_row(plane, y0 - 2 + dy, x0 - 3, rB);
        #pragma unroll
        for (int dx = 0; dx < 7; ++dx) {
            float v = wy0 * (wx0 * rA[dx] + wx1 * rA[dx + 1])
                    + wy1 * (wx0 * rB[dx] + wx1 * rB[dx + 1]);
            S[(dy * 7 + dx) * SSTRIDE + c] = v;
        }
        #pragma unroll
        for (int i = 0; i < 8; ++i) rA[i] = rB[i];
    }
}

// Per-position L2 norm over channels -> reciprocal scale (ref: /max(norm,1e-12)).
__device__ __forceinline__ void norms_from_S(int tid, const float* __restrict__ S,
                                             float* __restrict__ scaleS) {
    if (tid < P49) {
        float s = 0.0f;
        #pragma unroll 8
        for (int c = 0; c < C_DIM; ++c) {
            float v = S[tid * SSTRIDE + c];
            s += v * v;
        }
        scaleS[tid] = 1.0f / fmaxf(sqrtf(s), 1e-12f);
    }
}

// One 49x49 (padded 64x64) GEMM: A[row=pos][K=ch] x B[K=ch][col=pos] -> LDS D.
// Wave w owns column block nb = 16*w; fully unrolled over the 4 row blocks.
// All control flow here is wave-uniform => EXEC all-ones at every WMMA.
__device__ __forceinline__ void do_gemm(const _Float16* __restrict__ A,
                                        const _Float16* __restrict__ B,
                                        float* __restrict__ matD,
                                        int lane, int wave) {
    const int nb = wave << 4;
    // Preload the four B k-fragments once (nb is wave-invariant).
    v16h bf[4];
    #pragma unroll
    for (int kk = 0; kk < 4; ++kk) {
        v16h bv;
        const _Float16* bp = B + ((kk << 5) + ((lane >> 4) << 3)) * BSTRIDE + nb + (lane & 15);
        #pragma unroll
        for (int s = 0; s < 8; ++s) {
            bv[s]     = bp[s * BSTRIDE];
            bv[s + 8] = bp[(s + 16) * BSTRIDE];
        }
        bf[kk] = bv;
    }
    #pragma unroll
    for (int mt = 0; mt < 4; ++mt) {
        const int mb = mt << 4;
        // Preload all four A k-fragments, then run the WMMA chain.
        v16h af[4];
        #pragma unroll
        for (int kk = 0; kk < 4; ++kk) {
            union { v16h v; v8h h[2]; } au;
            const _Float16* ap = A + (mb + (lane & 15)) * ASTRIDE
                                   + (kk << 5) + ((lane >> 4) << 3);
            au.h[0] = *(const v8h*)(ap);
            au.h[1] = *(const v8h*)(ap + 16);
            af[kk] = au.v;
        }
        v8f acc = {};
        #pragma unroll
        for (int kk = 0; kk < 4; ++kk)
            acc = __builtin_amdgcn_wmma_f32_16x16x32_f16(
                false, af[kk], false, bf[kk], (short)0, acc, false, false);
        // Unmasked stage to padded LDS D (garbage rows/cols land in dead cells).
        const int col = nb + (lane & 15);
        const int rowBase = mb + ((lane >> 4) << 3);
        #pragma unroll
        for (int r = 0; r < 8; ++r)
            matD[(rowBase + r) * DSTRIDE + col] = acc[r];
    }
}

__global__ __launch_bounds__(128)
void glu_corr_kernel(const float* __restrict__ fmap_old,
                     const float* __restrict__ fmap_new,
                     const float* __restrict__ queries,
                     const float* __restrict__ pred_pos,
                     const float* __restrict__ anchors,
                     float* __restrict__ out) {
    __shared__ __align__(16) float     S[P49 * SSTRIDE];        // 25.9 KB staging
    __shared__ __align__(16) _Float16  matA[64 * ASTRIDE];      // 17.4 KB (a / anchors)
    __shared__ __align__(16) _Float16  matB[C_DIM * BSTRIDE];   // 18.4 KB (b, K-major)
    __shared__ float scaleS[P49];
    float* matD = S;   // D staging (64 x DSTRIDE f32 = 16.6 KB) aliases dead S

    const int n    = blockIdx.x;
    const int tid  = threadIdx.x;      // == channel c (128 threads)
    const int lane = tid & 31;
    const int wave = tid >> 5;
    const int c    = tid;

    const float qx = queries[2 * n],  qy = queries[2 * n + 1];
    const float px = pred_pos[2 * n], py = pred_pos[2 * n + 1];
    float* outn = out + (size_t)n * (2 * P49 * P49);
    const float* anc = anchors + (size_t)n * (C_DIM * P49);

    // ---- B = normalized features_cur (fmap_new @ pred_pos), K-major f16
    sample_to_S(fmap_new, px, py, c, S);
    __syncthreads();
    norms_from_S(tid, S, scaleS);
    __syncthreads();
    for (int p = 0; p < P49; ++p)
        matB[c * BSTRIDE + p] = (_Float16)(S[p * SSTRIDE + c] * scaleS[p]);
    __syncthreads();

    // ---- A = normalized features_last (fmap_old @ queries), row(pos)-major f16
    sample_to_S(fmap_old, qx, qy, c, S);
    __syncthreads();
    norms_from_S(tid, S, scaleS);
    __syncthreads();
    for (int p = 0; p < P49; ++p)
        matA[p * ASTRIDE + c] = (_Float16)(S[p * SSTRIDE + c] * scaleS[p]);
    __syncthreads();

    // Warm L2/L0 for the upcoming anchor stream while corr_st computes.
    __builtin_prefetch(anc + tid * P49, 0, 0);

    // ---- corr_st: gemm into LDS D, then coalesced stream-out of valid 49x49
    do_gemm(matA, matB, matD, lane, wave);
    __syncthreads();
    for (int i = tid; i < P49 * P49; i += 128) {
        int row = i / P49;
        int col = i - row * P49;
        outn[i] = matD[row * DSTRIDE + col];
    }
    __syncthreads();

    // ---- A <- normalized anchors (coalesced stream, transposed into [pos][ch])
    for (int i = tid; i < C_DIM * P49; i += 128) {
        int cc = i / P49;
        int p  = i - cc * P49;
        S[p * SSTRIDE + cc] = anc[i];
    }
    __syncthreads();
    norms_from_S(tid, S, scaleS);
    __syncthreads();
    for (int p = 0; p < P49; ++p)
        matA[p * ASTRIDE + c] = (_Float16)(S[p * SSTRIDE + c] * scaleS[p]);
    __syncthreads();

    // ---- corr_lt
    do_gemm(matA, matB, matD, lane, wave);
    __syncthreads();
    for (int i = tid; i < P49 * P49; i += 128) {
        int row = i / P49;
        int col = i - row * P49;
        outn[P49 * P49 + i] = matD[row * DSTRIDE + col];
    }
}

extern "C" void kernel_launch(void* const* d_in, const int* in_sizes, int n_in,
                              void* d_out, int out_size, void* d_ws, size_t ws_size,
                              hipStream_t stream) {
    const float* fmap_old = (const float*)d_in[0];
    const float* fmap_new = (const float*)d_in[1];
    const float* queries  = (const float*)d_in[2];
    const float* pred_pos = (const float*)d_in[3];
    const float* anchors  = (const float*)d_in[4];
    float* out = (float*)d_out;
    const int N = in_sizes[2] / 2;   // queries is (N,2)
    glu_corr_kernel<<<N, 128, 0, stream>>>(fmap_old, fmap_new, queries,
                                           pred_pos, anchors, out);
}
